// MM_Graph_FiBiNET_19439021981812
// MI455X (gfx1250) — compile-verified, matrix operand order checked by reading, not run on hardware
//
#include <hip/hip_runtime.h>
#include <hip/hip_bf16.h>
#include <math.h>

// ---------------- Problem constants ----------------
constexpr int Bn   = 16384;   // batch
constexpr int Ln   = 50;      // seq len
constexpr int Dn   = 128;     // embed dim
constexpr int Fn   = 6;       // fields
constexpr int Hn   = 4;       // heads
constexpr int HDn  = 32;      // head dim
constexpr int CINn = Fn * Dn + 15 * Dn;   // 2688
constexpr int H1n  = 512;
constexpr int H2n  = 256;

typedef __bf16 bf16_t;
typedef __attribute__((ext_vector_type(16))) __bf16 v16bf;
typedef __attribute__((ext_vector_type(8)))  float  v8f;

// ---------------- helpers ----------------
__device__ __forceinline__ v16bf load16bf(const bf16_t* p) {
  // two 16B chunks: elements [0..7] and [16..23] relative to p
  union { uint4 u[2]; v16bf v; } c;
  c.u[0] = *reinterpret_cast<const uint4*>(p);
  c.u[1] = *reinterpret_cast<const uint4*>(p + 16);
  return c.v;
}

__device__ __forceinline__ float gelu_exact(float v) {
  return 0.5f * v * (1.0f + erff(v * 0.70710678118654752f));
}

// ---------------- weight pack: f32 (K,N) row-major -> bf16 (N,K) ----------------
__global__ void k_transpose_bf16(const float* __restrict__ src, bf16_t* __restrict__ dst,
                                 int K, int N) {
  int idx = blockIdx.x * blockDim.x + threadIdx.x;
  if (idx >= K * N) return;
  int k = idx / N, n = idx % N;
  dst[(size_t)n * K + k] = (bf16_t)src[idx];
}

// f32 -> bf16 elementwise
__global__ void k_cvt_bf16(const float* __restrict__ src, bf16_t* __restrict__ dst, int n) {
  int idx = blockIdx.x * blockDim.x + threadIdx.x;
  if (idx < n) dst[idx] = (bf16_t)src[idx];
}

// ---------------- generic bf16 WMMA GEMM: C(M,N) = A(M,K) x Bt(N,K)^T ----------------
// Software-pipelined: loads for step k+1 are issued before the WMMAs of step k,
// so waits are partial and VMEM overlaps the matrix pipe.
// EPI 0: store f32.  EPI 1: +bias, BN, GELU -> bf16.  EPI 2: same -> f32.
template <int EPI>
__global__ __launch_bounds__(256) void k_gemm_bf16(
    const bf16_t* __restrict__ A, int lda,
    const bf16_t* __restrict__ Bt, int ldb,
    float* __restrict__ Cf, bf16_t* __restrict__ Cbf, int ldc,
    int M, int N, int K,
    const float* __restrict__ bias, const float* __restrict__ bng,
    const float* __restrict__ bnb) {
  const int lane = threadIdx.x & 31;
  const int wave = threadIdx.x >> 5;
  const int tilesN = N >> 6;                 // 64-wide N tiles per wave
  const int tiles  = (M >> 4) * tilesN;
  int t = blockIdx.x * 8 + wave;
  if (t >= tiles) return;
  const int tm = t / tilesN;
  const int tn = t % tilesN;
  const int half = lane >> 4;
  const int lm   = lane & 15;

  v8f acc[4] = {};
  const bf16_t* ap = A  + (size_t)(tm * 16 + lm) * lda + half * 8;
  const bf16_t* bp = Bt + (size_t)(tn * 64 + lm) * ldb + half * 8;

  // prologue: load k-step 0
  v16bf a_cur = load16bf(ap);
  v16bf b_cur0 = load16bf(bp);
  v16bf b_cur1 = load16bf(bp + (size_t)16 * ldb);
  v16bf b_cur2 = load16bf(bp + (size_t)32 * ldb);
  v16bf b_cur3 = load16bf(bp + (size_t)48 * ldb);

  for (int k0 = 32; k0 < K; k0 += 32) {
    // issue next step's loads first (overlaps with WMMAs below)
    v16bf a_nxt  = load16bf(ap + k0);
    v16bf b_nxt0 = load16bf(bp + k0);
    v16bf b_nxt1 = load16bf(bp + (size_t)16 * ldb + k0);
    v16bf b_nxt2 = load16bf(bp + (size_t)32 * ldb + k0);
    v16bf b_nxt3 = load16bf(bp + (size_t)48 * ldb + k0);

    acc[0] = __builtin_amdgcn_wmma_f32_16x16x32_bf16(
        false, a_cur, false, b_cur0, (short)0, acc[0], false, false);
    acc[1] = __builtin_amdgcn_wmma_f32_16x16x32_bf16(
        false, a_cur, false, b_cur1, (short)0, acc[1], false, false);
    acc[2] = __builtin_amdgcn_wmma_f32_16x16x32_bf16(
        false, a_cur, false, b_cur2, (short)0, acc[2], false, false);
    acc[3] = __builtin_amdgcn_wmma_f32_16x16x32_bf16(
        false, a_cur, false, b_cur3, (short)0, acc[3], false, false);

    a_cur = a_nxt;
    b_cur0 = b_nxt0; b_cur1 = b_nxt1; b_cur2 = b_nxt2; b_cur3 = b_nxt3;
  }

  // epilogue k-step
  acc[0] = __builtin_amdgcn_wmma_f32_16x16x32_bf16(
      false, a_cur, false, b_cur0, (short)0, acc[0], false, false);
  acc[1] = __builtin_amdgcn_wmma_f32_16x16x32_bf16(
      false, a_cur, false, b_cur1, (short)0, acc[1], false, false);
  acc[2] = __builtin_amdgcn_wmma_f32_16x16x32_bf16(
      false, a_cur, false, b_cur2, (short)0, acc[2], false, false);
  acc[3] = __builtin_amdgcn_wmma_f32_16x16x32_bf16(
      false, a_cur, false, b_cur3, (short)0, acc[3], false, false);

#pragma unroll
  for (int j = 0; j < 4; ++j) {
    int n = tn * 64 + j * 16 + lm;
#pragma unroll
    for (int r = 0; r < 8; ++r) {
      int m = tm * 16 + half * 8 + r;
      float v = acc[j][r];
      if (EPI == 0) {
        Cf[(size_t)m * ldc + n] = v;
      } else {
        v += bias[n];
        v = v * (bng[n] * 0.9999950000374997f) + bnb[n];  // g / sqrt(1+1e-5)
        v = gelu_exact(v);
        if (EPI == 1) Cbf[(size_t)m * ldc + n] = (bf16_t)v;
        else          Cf [(size_t)m * ldc + n] = v;
      }
    }
  }
}

// ---------------- feature builder: X[b, 6, 128] (f32 + bf16) ----------------
__global__ __launch_bounds__(128) void k_features(
    const int* __restrict__ item_id, const int* __restrict__ likes,
    const int* __restrict__ views, const int* __restrict__ item_seq,
    const float* __restrict__ item_emb, const float* __restrict__ cate_emb,
    const float* __restrict__ Y, const float* __restrict__ mm_b,
    const float* __restrict__ ln_g, const float* __restrict__ ln_b,
    float* __restrict__ Xf, bf16_t* __restrict__ Xbf) {
  const int b = blockIdx.x;
  const int d = threadIdx.x;  // 128 threads
  __shared__ float red[128];

  // img feature: LayerNorm(Y + mm_b) -> GELU
  float y = Y[(size_t)b * Dn + d] + mm_b[d];
  red[d] = y; __syncthreads();
  for (int s = 64; s > 0; s >>= 1) { if (d < s) red[d] += red[d + s]; __syncthreads(); }
  float mu = red[0] * (1.0f / 128.0f);
  __syncthreads();
  float cd = y - mu;
  red[d] = cd * cd; __syncthreads();
  for (int s = 64; s > 0; s >>= 1) { if (d < s) red[d] += red[d + s]; __syncthreads(); }
  float var = red[0] * (1.0f / 128.0f);
  float yn  = cd * rsqrtf(var + 1e-5f) * ln_g[d] + ln_b[d];
  float img = gelu_exact(yn);

  // hist feature: masked mean over sequence
  float acc = 0.0f; int cnt = 0;
  const int* seq = item_seq + (size_t)b * Ln;
  for (int l = 0; l < Ln; ++l) {
    int v = seq[l];
    if (v != 0) { acc += item_emb[(size_t)v * Dn + d]; ++cnt; }
  }
  float hist = acc / (float)(cnt < 1 ? 1 : cnt);

  float likef = cate_emb[(size_t)likes[b] * Dn + d];
  float viewf = cate_emb[(size_t)views[b] * Dn + d];
  float idf   = item_emb[(size_t)item_id[b] * Dn + d];

  size_t base = (size_t)b * (Fn * Dn);
  float vals[Fn] = {0.0f, likef, viewf, idf, img, hist};
#pragma unroll
  for (int n = 0; n < Fn; ++n) {
    Xf [base + n * Dn + d] = vals[n];
    Xbf[base + n * Dn + d] = (bf16_t)vals[n];
  }
}

// ---------------- GAT attention + ELU + SE; s overwrites X ----------------
__global__ __launch_bounds__(128) void k_attn_se(
    const float* __restrict__ HP, float* __restrict__ Xf, bf16_t* __restrict__ Xbf,
    const float* __restrict__ gnn_a, const float* __restrict__ se_w1,
    const float* __restrict__ se_b1, const float* __restrict__ se_w2,
    const float* __restrict__ se_b2, bf16_t* __restrict__ Cmat) {
  const int b = blockIdx.x;
  const int t = threadIdx.x;        // 128 threads: head h = wave, o = lane
  const int lane = t & 31;
  const int h = t >> 5;

  __shared__ float sh_hp[Fn][Dn];
  __shared__ float sh_gf[Fn][Dn];
  __shared__ float sh_ei[Hn][Fn], sh_ej[Hn][Fn];
  __shared__ float sh_att[Hn][Fn][Fn];
  __shared__ float sh_red[Hn];
  __shared__ float sh_z[Fn], sh_r1[3], sh_w[Fn];

  const size_t base = (size_t)b * (Fn * Dn);
  float xv[Fn];
#pragma unroll
  for (int n = 0; n < Fn; ++n) {
    sh_hp[n][t] = HP[base + n * Dn + t];
    xv[n] = Xf[base + n * Dn + t];
  }
  __syncthreads();

  float a1v = gnn_a[h * 2 * HDn + lane];
  float a2v = gnn_a[h * 2 * HDn + HDn + lane];
#pragma unroll
  for (int n = 0; n < Fn; ++n) {
    float v1 = sh_hp[n][h * HDn + lane] * a1v;
    float v2 = sh_hp[n][h * HDn + lane] * a2v;
#pragma unroll
    for (int off = 16; off > 0; off >>= 1) {
      v1 += __shfl_xor(v1, off, 32);
      v2 += __shfl_xor(v2, off, 32);
    }
    if (lane == 0) { sh_ei[h][n] = v1; sh_ej[h][n] = v2; }
  }
  __syncthreads();

  if (t < Hn * Fn) {
    int hh = t / Fn, n = t % Fn;
    float e[Fn], mx = -3.0e38f;
#pragma unroll
    for (int m = 0; m < Fn; ++m) {
      float x = sh_ei[hh][n] + sh_ej[hh][m];
      e[m] = x > 0.0f ? x : 0.2f * x;          // leaky relu
      mx = fmaxf(mx, e[m]);
    }
    float s = 0.0f;
#pragma unroll
    for (int m = 0; m < Fn; ++m) { e[m] = expf(e[m] - mx); s += e[m]; }
    float inv = 1.0f / s;
#pragma unroll
    for (int m = 0; m < Fn; ++m) sh_att[hh][n][m] = e[m] * inv;
  }
  __syncthreads();

#pragma unroll
  for (int n = 0; n < Fn; ++n) {
    float hn = 0.0f;
#pragma unroll
    for (int m = 0; m < Fn; ++m) hn += sh_att[h][n][m] * sh_hp[m][t];
    float g  = hn + xv[n];
    float gf = g > 0.0f ? g : expm1f(g);       // ELU
    sh_gf[n][t] = gf;
    Cmat[(size_t)b * CINn + n * Dn + t] = (bf16_t)gf;  // first 768 cols of c
  }
  __syncthreads();

  // z[n] = mean_d gnn_feat
  for (int n = 0; n < Fn; ++n) {
    float v = sh_gf[n][t];
#pragma unroll
    for (int off = 16; off > 0; off >>= 1) v += __shfl_xor(v, off, 32);
    if (lane == 0) sh_red[h] = v;
    __syncthreads();
    if (t == 0) sh_z[n] = (sh_red[0] + sh_red[1] + sh_red[2] + sh_red[3]) * (1.0f / 128.0f);
    __syncthreads();
  }

  // SE block
  if (t < 3) {
    float a = se_b1[t];
#pragma unroll
    for (int n = 0; n < Fn; ++n) a += sh_z[n] * se_w1[n * 3 + t];
    sh_r1[t] = fmaxf(a, 0.0f);
  }
  __syncthreads();
  if (t < Fn) {
    float a = se_b2[t];
#pragma unroll
    for (int j = 0; j < 3; ++j) a += sh_r1[j] * se_w2[j * Fn + t];
    sh_w[t] = 1.0f / (1.0f + expf(-a));
  }
  __syncthreads();

#pragma unroll
  for (int n = 0; n < Fn; ++n) {
    float s = sh_gf[n][t] * sh_w[n];
    Xf [base + n * Dn + t] = s;                // s overwrites x (f32)
    Xbf[base + n * Dn + t] = (bf16_t)s;        // bf16 operand for bilinear GEMM
  }
}

// ---------------- pairwise products -> Cmat[:, 768:] ----------------
__global__ void k_pairs(const float* __restrict__ VID, const float* __restrict__ S,
                        bf16_t* __restrict__ Cmat) {
  int idx = blockIdx.x * blockDim.x + threadIdx.x;
  const int total = Bn * 15 * Dn;
  if (idx >= total) return;
  int d = idx & (Dn - 1);
  int rest = idx >> 7;
  int p = rest % 15;
  int b = rest / 15;
  static const int pi[15] = {0,0,0,0,0, 1,1,1,1, 2,2,2, 3,3, 4};
  static const int pj[15] = {1,2,3,4,5, 2,3,4,5, 3,4,5, 4,5, 5};
  float v = VID[(size_t)b * (5 * Dn) + pi[p] * Dn + d] *
            S  [(size_t)b * (Fn * Dn) + pj[p] * Dn + d];
  Cmat[(size_t)b * CINn + Fn * Dn + p * Dn + d] = (bf16_t)v;
}

// ---------------- final dot + sigmoid ----------------
__global__ __launch_bounds__(256) void k_final(const float* __restrict__ H2,
                                               const float* __restrict__ w3,
                                               const float* __restrict__ b3,
                                               float* __restrict__ out) {
  int b = blockIdx.x, t = threadIdx.x;
  __shared__ float red[256];
  red[t] = H2[(size_t)b * H2n + t] * w3[t];
  __syncthreads();
  for (int s = 128; s > 0; s >>= 1) { if (t < s) red[t] += red[t + s]; __syncthreads(); }
  if (t == 0) out[b] = 1.0f / (1.0f + expf(-(red[0] + b3[0])));
}

// ---------------- host launch ----------------
extern "C" void kernel_launch(void* const* d_in, const int* in_sizes, int n_in,
                              void* d_out, int out_size, void* d_ws, size_t ws_size,
                              hipStream_t stream) {
  const int*   item_id  = (const int*)  d_in[0];
  const float* item_mm  = (const float*)d_in[1];
  const int*   likes    = (const int*)  d_in[2];
  const int*   views    = (const int*)  d_in[3];
  const int*   item_seq = (const int*)  d_in[4];
  const float* item_emb = (const float*)d_in[5];
  const float* cate_emb = (const float*)d_in[6];
  const float* mm_w     = (const float*)d_in[7];
  const float* mm_b     = (const float*)d_in[8];
  const float* ln_g     = (const float*)d_in[9];
  const float* ln_b     = (const float*)d_in[10];
  const float* gnn_W    = (const float*)d_in[11];
  const float* gnn_a    = (const float*)d_in[12];
  const float* se_w1    = (const float*)d_in[13];
  const float* se_b1    = (const float*)d_in[14];
  const float* se_w2    = (const float*)d_in[15];
  const float* se_b2    = (const float*)d_in[16];
  const float* bi_W     = (const float*)d_in[17];
  const float* mlp_w1   = (const float*)d_in[18];
  const float* mlp_b1   = (const float*)d_in[19];
  const float* bn1_g    = (const float*)d_in[20];
  const float* bn1_b    = (const float*)d_in[21];
  const float* mlp_w2   = (const float*)d_in[22];
  const float* mlp_b2   = (const float*)d_in[23];
  const float* bn2_g    = (const float*)d_in[24];
  const float* bn2_b    = (const float*)d_in[25];
  const float* mlp_w3   = (const float*)d_in[26];
  const float* mlp_b3   = (const float*)d_in[27];
  float* out = (float*)d_out;

  // ---- workspace layout ----
  size_t off = 0;
  auto alloc = [&](size_t bytes) -> void* {
    void* p = (char*)d_ws + off;
    off += (bytes + 255) & ~(size_t)255;
    return p;
  };
  bf16_t* w_mm_t  = (bf16_t*)alloc((size_t)Dn * Dn * 2);
  bf16_t* w_gnn_t = (bf16_t*)alloc((size_t)Dn * Dn * 2);
  bf16_t* w_bi_t  = (bf16_t*)alloc((size_t)5 * Dn * Dn * 2);
  bf16_t* w1_t    = (bf16_t*)alloc((size_t)H1n * CINn * 2);
  bf16_t* w2_t    = (bf16_t*)alloc((size_t)H2n * H1n * 2);
  bf16_t* MMbf    = (bf16_t*)alloc((size_t)Bn * Dn * 2);
  float*  Y       = (float*) alloc((size_t)Bn * Dn * 4);
  float*  Xf      = (float*) alloc((size_t)Bn * Fn * Dn * 4);  // later: s (f32)
  bf16_t* Xbf     = (bf16_t*)alloc((size_t)Bn * Fn * Dn * 2);  // later: s (bf16)
  float*  HP      = (float*) alloc((size_t)Bn * Fn * Dn * 4);
  float*  VID     = (float*) alloc((size_t)Bn * 5 * Dn * 4);
  bf16_t* Cmat    = (bf16_t*)alloc((size_t)Bn * CINn * 2);
  bf16_t* H1      = (bf16_t*)alloc((size_t)Bn * H1n * 2);
  float*  H2      = (float*) alloc((size_t)Bn * H2n * 4);
  (void)ws_size; (void)n_in; (void)in_sizes; (void)out_size;

  const int TPB = 256;
  auto blk = [](int n, int t) { return (n + t - 1) / t; };

  // ---- pack weights (bf16, N-major) ----
  k_transpose_bf16<<<blk(Dn * Dn, TPB), TPB, 0, stream>>>(mm_w, w_mm_t, Dn, Dn);
  for (int h = 0; h < Hn; ++h)  // gnn_W (H,D,HD) -> cols [h*32, h*32+32)
    k_transpose_bf16<<<blk(Dn * HDn, TPB), TPB, 0, stream>>>(
        gnn_W + (size_t)h * Dn * HDn, w_gnn_t + (size_t)h * HDn * Dn, Dn, HDn);
  for (int f = 0; f < 5; ++f)
    k_transpose_bf16<<<blk(Dn * Dn, TPB), TPB, 0, stream>>>(
        bi_W + (size_t)f * Dn * Dn, w_bi_t + (size_t)f * Dn * Dn, Dn, Dn);
  k_transpose_bf16<<<blk(CINn * H1n, TPB), TPB, 0, stream>>>(mlp_w1, w1_t, CINn, H1n);
  k_transpose_bf16<<<blk(H1n * H2n, TPB), TPB, 0, stream>>>(mlp_w2, w2_t, H1n, H2n);
  k_cvt_bf16<<<blk(Bn * Dn, TPB), TPB, 0, stream>>>(item_mm, MMbf, Bn * Dn);

  // ---- Y = item_mm @ mm_w  (pre-LN) ----
  {
    int tiles = (Bn / 16) * (Dn / 64);
    k_gemm_bf16<0><<<blk(tiles, 8), TPB, 0, stream>>>(
        MMbf, Dn, w_mm_t, Dn, Y, nullptr, Dn, Bn, Dn, Dn, nullptr, nullptr, nullptr);
  }

  // ---- feature assembly ----
  k_features<<<Bn, 128, 0, stream>>>(item_id, likes, views, item_seq, item_emb,
                                     cate_emb, Y, mm_b, ln_g, ln_b, Xf, Xbf);

  // ---- HP = X @ gnn_W  (M = B*6) ----
  {
    int tiles = (Bn * Fn / 16) * (Dn / 64);
    k_gemm_bf16<0><<<blk(tiles, 8), TPB, 0, stream>>>(
        Xbf, Dn, w_gnn_t, Dn, HP, nullptr, Dn, Bn * Fn, Dn, Dn,
        nullptr, nullptr, nullptr);
  }

  // ---- attention + SE (writes gnn_feat -> Cmat[:, :768], s -> Xf/Xbf) ----
  k_attn_se<<<Bn, 128, 0, stream>>>(HP, Xf, Xbf, gnn_a, se_w1, se_b1, se_w2,
                                    se_b2, Cmat);

  // ---- bilinear: VID[:, f] = s[:, f] @ bi_W[f] ----
  for (int f = 0; f < 5; ++f) {
    int tiles = (Bn / 16) * (Dn / 64);
    k_gemm_bf16<0><<<blk(tiles, 8), TPB, 0, stream>>>(
        Xbf + (size_t)f * Dn, Fn * Dn, w_bi_t + (size_t)f * Dn * Dn, Dn,
        VID + (size_t)f * Dn, nullptr, 5 * Dn, Bn, Dn, Dn,
        nullptr, nullptr, nullptr);
  }

  // ---- pairwise products -> Cmat[:, 768:] ----
  k_pairs<<<blk(Bn * 15 * Dn, TPB), TPB, 0, stream>>>(VID, Xf, Cmat);

  // ---- MLP1: c @ w1 + b1, BN, GELU -> H1 (bf16) ----
  {
    int tiles = (Bn / 16) * (H1n / 64);
    k_gemm_bf16<1><<<blk(tiles, 8), TPB, 0, stream>>>(
        Cmat, CINn, w1_t, CINn, nullptr, H1, H1n, Bn, H1n, CINn,
        mlp_b1, bn1_g, bn1_b);
  }
  // ---- MLP2: h1 @ w2 + b2, BN, GELU -> H2 (f32) ----
  {
    int tiles = (Bn / 16) * (H2n / 64);
    k_gemm_bf16<2><<<blk(tiles, 8), TPB, 0, stream>>>(
        H1, H1n, w2_t, H1n, H2, nullptr, H2n, Bn, H2n, H1n,
        mlp_b2, bn2_g, bn2_b);
  }
  // ---- final dot + sigmoid ----
  k_final<<<Bn, 256, 0, stream>>>(H2, mlp_w3, mlp_b3, out);
}
